// RechitsGCN_11493332484802
// MI455X (gfx1250) — compile-verified, matrix operand order checked by reading, not run on hardware
//
#include <hip/hip_runtime.h>
#include <hip/hip_bf16.h>

typedef __attribute__((ext_vector_type(16))) _Float16 v16h;
typedef __attribute__((ext_vector_type(8)))  _Float16 v8h;
typedef __attribute__((ext_vector_type(8)))  float    v8f;
typedef __attribute__((ext_vector_type(4)))  unsigned int u32x4;
typedef __attribute__((ext_vector_type(8)))  int      i32x8;
typedef __attribute__((ext_vector_type(4)))  int      i32x4;
typedef __attribute__((address_space(3)))    char     lds_char_t;

#define WMMA_F16(a, b, c) \
    __builtin_amdgcn_wmma_f32_16x16x32_f16(false, (a), false, (b), (short)0, (c), false, false)

// CDNA5 16-bit A/B operand layout (ISA 7.12.2): for lane L (laneHi = L>=16),
// the 16 halves hold K = {8*laneHi + 0..7} then {16 + 8*laneHi + 0..7}.
// For a row-major K-contiguous row in LDS this is two 16-byte ds_load_b128.
__device__ __forceinline__ v16h ld_ab32(const _Float16* p /* = row + 8*laneHi */) {
    v8h lo = *(const v8h*)(p);
    v8h hi = *(const v8h*)(p + 16);
    return __builtin_shufflevector(lo, hi, 0, 1, 2, 3, 4, 5, 6, 7,
                                           8, 9, 10, 11, 12, 13, 14, 15);
}
// K padded 16->32 with zeros: only the low 16 K are valid.
__device__ __forceinline__ v16h ld_ab16(const _Float16* p /* = row + 8*laneHi */) {
    v8h lo = *(const v8h*)(p);
    v8h z  = {};
    return __builtin_shufflevector(lo, z, 0, 1, 2, 3, 4, 5, 6, 7,
                                          8, 9, 10, 11, 12, 13, 14, 15);
}

// Problem constants
#define NB    64
#define NCL   64
#define NR    128
#define D_IN  16
#define D_OUT 32

__global__ __launch_bounds__(256)
void rechits_gcn_fused(const float* __restrict__ xg,
                       const float* __restrict__ mg,
                       const float* __restrict__ Qg,
                       const float* __restrict__ Kg,
                       const float* __restrict__ Vg,
                       const float* __restrict__ W1g,
                       const float* __restrict__ b1g,
                       const float* __restrict__ W2g,
                       const float* __restrict__ b2g,
                       const float* __restrict__ gmg,
                       const float* __restrict__ btg,
                       float* __restrict__ out)
{
    // ---- LDS: persistent region + phase-union region (47104 B total) ----
    __shared__ __align__(16) char s_per[14336];
    __shared__ __align__(16) char s_uni[32768];

    float*    mask_s = (float*)(s_per + 0);          // 128 f32
    _Float16* W1t    = (_Float16*)(s_per + 512);     // 32x32 f16, TRANSPOSED: W1t[f][e]
    _Float16* W2t    = (_Float16*)(s_per + 2560);    // 32x32 f16, TRANSPOSED: W2t[f][e]
    float*    b1s    = (float*)(s_per + 4608);       // 32 f32
    float*    b2s    = (float*)(s_per + 4736);       // 32 f32
    float*    gms    = (float*)(s_per + 4864);       // 32 f32
    float*    bts    = (float*)(s_per + 4992);       // 32 f32
    _Float16* vt     = (_Float16*)(s_per + 5120);    // 32x128 f16, TRANSPOSED: vt[e][row]
    float*    wsum   = (float*)(s_per + 13312);      // 8 waves x 32 f32 partial pools

    // Phase 1 union view: x (f32, TDM-staged), q, k, transposed weight tiles
    float*    xf = (float*)(s_uni + 0);              // 128x16 f32 (TDM destination)
    _Float16* qh = (_Float16*)(s_uni + 8192);        // 128x32 row-major
    _Float16* kh = (_Float16*)(s_uni + 16384);       // 128x32 row-major
    _Float16* Qt = (_Float16*)(s_uni + 24576);       // 32x16 TRANSPOSED: Qt[e][d]
    _Float16* Kt = (_Float16*)(s_uni + 25600);       // 32x16 TRANSPOSED
    _Float16* Vt = (_Float16*)(s_uni + 26624);       // 32x16 TRANSPOSED
    // Phase 2 union view
    _Float16* attn = (_Float16*)(s_uni + 0);         // 128x128 row-major
    // Phase 3 union view
    _Float16* sah = (_Float16*)(s_uni + 0);          // 128x32 row-major
    _Float16* hh  = (_Float16*)(s_uni + 8192);       // 128x32 row-major

    const int tid    = threadIdx.x;
    const int lane   = tid & 31;
    const int wave   = tid >> 5;          // 0..7, owns rows [16w, 16w+16)
    const int laneHi = (lane >> 4) & 1;
    const int ln     = lane & 15;
    const int m0     = wave * 16;
    const int koff   = 8 * laneHi;        // per-lane K phase offset for operand loads
    const int bc     = blockIdx.x;        // cluster index in [0, B*NCL)

    // ---- TDM: DMA the x tile (128x16 f32, 8KB) into LDS via tensor_load_to_lds ----
    if (wave == 0) {
        // LDS-relative byte offset of the staging region (hardware adds LDS_BASE)
        unsigned ldsOff = (unsigned)(unsigned long long)(lds_char_t*)s_uni;
        unsigned long long ga = (unsigned long long)(xg + (size_t)bc * (NR * D_IN));
        // D# group 0: count=1 | lds_addr | global_addr[56:0] | type=2
        u32x4 g0 = { 1u,
                     ldsOff,
                     (unsigned)(ga & 0xFFFFFFFFu),
                     (unsigned)((ga >> 32) & 0x01FFFFFFu) | (2u << 30) };
        // D# group 1: data_size=4B; 1-D tile of 2048 elements; stride 2048
        i32x8 g1 = { (int)(2u << 16),                  // data_size = 2 (4 bytes)
                     (int)((2048u & 0xFFFFu) << 16),   // tensor_dim0[15:0]
                     (int)(1u << 16),                  // tensor_dim0[31:16]=0 | tensor_dim1=1
                     (int)(2048u << 16),               // tensor_dim1 hi=0 | tile_dim0=2048
                     1,                                // tile_dim1=1 | tile_dim2=0
                     2048,                             // tensor_dim0_stride[31:0]
                     0,                                // stride hi | tensor_dim1_stride lo
                     0 };
        i32x4 gz4 = { 0, 0, 0, 0 };                    // <=2-D tensor: groups 2/3 unused
        i32x8 gz8 = { 0, 0, 0, 0, 0, 0, 0, 0 };
        __builtin_amdgcn_tensor_load_to_lds(g0, g1, gz4, gz4, gz8, 0);
        __builtin_amdgcn_s_wait_tensorcnt(0);
    }

    // ---- Stage small params into LDS (weights transposed for B-operand contiguity) ----
    {
        for (int i = tid; i < NR; i += 256) mask_s[i] = mg[(size_t)bc * NR + i];
        for (int i = tid; i < D_IN * D_OUT; i += 256) {
            int e = i >> 4, d = i & 15;               // Qt[e*16+d] = Q[d][e]
            Qt[i] = (_Float16)Qg[d * D_OUT + e];
            Kt[i] = (_Float16)Kg[d * D_OUT + e];
            Vt[i] = (_Float16)Vg[d * D_OUT + e];
        }
        for (int i = tid; i < D_OUT * D_OUT; i += 256) {
            int f = i >> 5, e = i & 31;               // W1t[f*32+e] = W1[e][f]
            W1t[i] = (_Float16)W1g[e * D_OUT + f];
            W2t[i] = (_Float16)W2g[e * D_OUT + f];
        }
        if (tid < D_OUT) {
            b1s[tid] = b1g[tid];
            b2s[tid] = b2g[tid];
            gms[tid] = gmg[tid];
            bts[tid] = btg[tid];
        }
    }
    __syncthreads();   // TDM tile + params visible to all waves

    // ---- Phase 1: q/k/v projections (K = D_IN = 16, zero-padded to 32) ----
    // Build A operand from the f32 staging tile with inline f32->f16 conversion.
    v16h ax;
    {
        const float* xrow = xf + (m0 + ln) * D_IN + koff;
        v8h lo;
        #pragma unroll
        for (int j = 0; j < 8; ++j) lo[j] = (_Float16)xrow[j];
        v8h z = {};
        ax = __builtin_shufflevector(lo, z, 0, 1, 2, 3, 4, 5, 6, 7,
                                            8, 9, 10, 11, 12, 13, 14, 15);
    }
    #pragma unroll
    for (int t = 0; t < 2; ++t) {
        v16h bq = ld_ab16(Qt + (16 * t + ln) * D_IN + koff);
        v16h bk = ld_ab16(Kt + (16 * t + ln) * D_IN + koff);
        v16h bv = ld_ab16(Vt + (16 * t + ln) * D_IN + koff);
        v8f zc = {};
        v8f aq = WMMA_F16(ax, bq, zc);
        v8f ak = WMMA_F16(ax, bk, zc);
        v8f av = WMMA_F16(ax, bv, zc);
        #pragma unroll
        for (int r = 0; r < 8; ++r) {
            int row = m0 + r + 8 * laneHi;
            int col = 16 * t + ln;
            qh[row * D_OUT + col] = (_Float16)aq[r];
            kh[row * D_OUT + col] = (_Float16)ak[r];
            vt[col * NR + row]    = (_Float16)av[r];   // transposed store for sa B-operand
        }
    }
    __syncthreads();   // k/v visible cluster-wide

    // ---- Phase 2: logits = q k^T * scale + mask bias; register softmax ----
    v16h aqm = ld_ab32(qh + (m0 + ln) * D_OUT + koff);
    v8f la[8];
    #pragma unroll
    for (int t = 0; t < 8; ++t) {
        v16h bkt = ld_ab32(kh + (16 * t + ln) * D_OUT + koff);  // B[e][n] = k[16t+n][e]
        v8f zc = {};
        la[t] = WMMA_F16(aqm, bkt, zc);
    }
    const float scale = 0.17677669529663687f;  // 1/sqrt(32)
    float rmax[8], rsum[8];
    #pragma unroll
    for (int r = 0; r < 8; ++r) rmax[r] = -3.0e38f;
    #pragma unroll
    for (int t = 0; t < 8; ++t) {
        float bias = (1.0f - mask_s[16 * t + ln]) * -1.0e9f;
        #pragma unroll
        for (int r = 0; r < 8; ++r) {
            la[t][r] = la[t][r] * scale + bias;
            rmax[r]  = fmaxf(rmax[r], la[t][r]);
        }
    }
    #pragma unroll
    for (int r = 0; r < 8; ++r) {
        #pragma unroll
        for (int off = 1; off < 16; off <<= 1)
            rmax[r] = fmaxf(rmax[r], __shfl_xor(rmax[r], off, 32));
        rsum[r] = 0.0f;
    }
    #pragma unroll
    for (int t = 0; t < 8; ++t) {
        #pragma unroll
        for (int r = 0; r < 8; ++r) {
            la[t][r] = __expf(la[t][r] - rmax[r]);
            rsum[r] += la[t][r];
        }
    }
    #pragma unroll
    for (int r = 0; r < 8; ++r) {
        #pragma unroll
        for (int off = 1; off < 16; off <<= 1)
            rsum[r] += __shfl_xor(rsum[r], off, 32);
        rsum[r] = 1.0f / rsum[r];
    }
    __syncthreads();   // x/q/k region dead everywhere -> reuse union as attn
    #pragma unroll
    for (int t = 0; t < 8; ++t) {
        #pragma unroll
        for (int r = 0; r < 8; ++r) {
            int row = m0 + r + 8 * laneHi;
            attn[row * NR + 16 * t + ln] = (_Float16)(la[t][r] * rsum[r]);
        }
    }
    __syncthreads();

    // ---- Phase 3: sa = attn @ v (4 K-tiles chained into the accumulators) ----
    v8f sa0 = {}, sa1 = {};
    #pragma unroll
    for (int j = 0; j < 4; ++j) {
        v16h aA  = ld_ab32(attn + (m0 + ln) * NR + 32 * j + koff);
        v16h bv0 = ld_ab32(vt + ln * NR + 32 * j + koff);          // B[kk][n] = v[32j+kk][n]
        v16h bv1 = ld_ab32(vt + (16 + ln) * NR + 32 * j + koff);
        sa0 = WMMA_F16(aA, bv0, sa0);
        sa1 = WMMA_F16(aA, bv1, sa1);
    }
    __syncthreads();   // attn region dead -> reuse union as sah/hh
    #pragma unroll
    for (int r = 0; r < 8; ++r) {
        int row = m0 + r + 8 * laneHi;
        sah[row * D_OUT + ln]      = (_Float16)sa0[r];
        sah[row * D_OUT + 16 + ln] = (_Float16)sa1[r];
    }
    __syncthreads();

    // ---- Phase 4: h = relu(sa W1 + b1), d = relu(h W2 + b2) ----
    v16h aS = ld_ab32(sah + (m0 + ln) * D_OUT + koff);
    #pragma unroll
    for (int t = 0; t < 2; ++t) {
        v16h bW = ld_ab32(W1t + (16 * t + ln) * D_OUT + koff);     // B[e][f] = W1[e][16t+f]
        v8f zc = {};
        v8f hv = WMMA_F16(aS, bW, zc);
        float b = b1s[16 * t + ln];
        #pragma unroll
        for (int r = 0; r < 8; ++r) {
            int row = m0 + r + 8 * laneHi;
            hh[row * D_OUT + 16 * t + ln] = (_Float16)fmaxf(hv[r] + b, 0.0f);
        }
    }
    __syncthreads();
    v16h aH = ld_ab32(hh + (m0 + ln) * D_OUT + koff);
    v8f rr[2];
    #pragma unroll
    for (int t = 0; t < 2; ++t) {
        v16h bW = ld_ab32(W2t + (16 * t + ln) * D_OUT + koff);
        v8f zc = {};
        v8f dv = WMMA_F16(aH, bW, zc);
        float b = b2s[16 * t + ln];
        v8f sat = (t == 0) ? sa0 : sa1;
        #pragma unroll
        for (int r = 0; r < 8; ++r)
            rr[t][r] = fmaxf(dv[r] + b, 0.0f) + sat[r];   // residual add in-register
    }

    // ---- Phase 5: LayerNorm over D_OUT per row + mask; Phase 6: pooling ----
    float g0v = gms[ln], g1v = gms[16 + ln];
    float be0 = bts[ln], be1 = bts[16 + ln];
    float p0 = 0.0f, p1 = 0.0f;
    #pragma unroll
    for (int r = 0; r < 8; ++r) {
        float s = rr[0][r] + rr[1][r];
        #pragma unroll
        for (int off = 1; off < 16; off <<= 1)
            s += __shfl_xor(s, off, 32);
        float mu = s * (1.0f / 32.0f);
        float d0 = rr[0][r] - mu, d1 = rr[1][r] - mu;
        float ss = d0 * d0 + d1 * d1;
        #pragma unroll
        for (int off = 1; off < 16; off <<= 1)
            ss += __shfl_xor(ss, off, 32);
        float inv = rsqrtf(ss * (1.0f / 32.0f) + 1.0e-3f);
        int row = m0 + r + 8 * laneHi;
        float mrow = mask_s[row];
        p0 += (d0 * inv * g0v + be0) * mrow;
        p1 += (d1 * inv * g1v + be1) * mrow;
    }
    p0 += __shfl_xor(p0, 16, 32);   // fold the two 8-row halves of the strip
    p1 += __shfl_xor(p1, 16, 32);
    if (!laneHi) {
        wsum[wave * D_OUT + ln]      = p0;
        wsum[wave * D_OUT + 16 + ln] = p1;
    }
    __syncthreads();

    if (tid < D_OUT) {
        float s = 0.0f;
        #pragma unroll
        for (int w = 0; w < 8; ++w) s += wsum[w * D_OUT + tid];
        float n = 0.0f;
        for (int i = 0; i < NR; ++i) n += mask_s[i];
        out[(size_t)bc * D_OUT + tid] = (n > 0.0f) ? (s / n) : 0.0f;
    }
}

extern "C" void kernel_launch(void* const* d_in, const int* in_sizes, int n_in,
                              void* d_out, int out_size, void* d_ws, size_t ws_size,
                              hipStream_t stream) {
    const float* x    = (const float*)d_in[0];
    const float* mask = (const float*)d_in[1];
    const float* Q    = (const float*)d_in[2];
    const float* K    = (const float*)d_in[3];
    const float* V    = (const float*)d_in[4];
    const float* W1   = (const float*)d_in[5];
    const float* b1   = (const float*)d_in[6];
    const float* W2   = (const float*)d_in[7];
    const float* b2   = (const float*)d_in[8];
    const float* gm   = (const float*)d_in[9];
    const float* bt   = (const float*)d_in[10];
    float* out = (float*)d_out;

    dim3 grid(NB * NCL);   // one workgroup (8 wave32) per cluster
    dim3 block(256);
    rechits_gcn_fused<<<grid, block, 0, stream>>>(x, mask, Q, K, V, W1, b1, W2, b2, gm, bt, out);
}
